// CapsNetFeatureExtractor_20194936226664
// MI455X (gfx1250) — matrix operand order repaired and checked by reading, not compiled, not run on hardware
//
#include <hip/hip_runtime.h>
#include <cstdint>
#include <cstddef>

// ---------------- types ----------------
typedef __bf16 bf16;
typedef __bf16 v16bf __attribute__((ext_vector_type(16)));
typedef __bf16 v8bf  __attribute__((ext_vector_type(8)));
typedef float  v8f   __attribute__((ext_vector_type(8)));

#ifndef __has_builtin
#define __has_builtin(x) 0
#endif
#if __has_builtin(__builtin_amdgcn_global_load_async_to_lds_b32)
#define HAS_ASYNC_LDS 1
typedef __attribute__((address_space(1))) int gint;   // global (device) int
typedef __attribute__((address_space(3))) int lint;   // LDS int
#else
#define HAS_ASYNC_LDS 0
#endif

static __device__ inline void wait_async_zero() {
#if HAS_ASYNC_LDS
#if __has_builtin(__builtin_amdgcn_s_wait_asynccnt)
  __builtin_amdgcn_s_wait_asynccnt(0);
#else
  asm volatile("s_wait_asynccnt 0" ::: "memory");
#endif
#endif
}

// ---------------- problem constants ----------------
static constexpr int BATCH   = 64;
static constexpr int IN_C    = 3;
static constexpr int IMG     = 64;
static constexpr int C1      = 256;   // conv1 out channels
static constexpr int S1      = 56;    // conv1 spatial
static constexpr int C2      = 256;   // pconv out channels (NC*PC_DIM)
static constexpr int PRIM    = 24;    // pconv spatial
static constexpr int ROUTES  = 4608;  // 8*24*24
static constexpr int PC_DIM  = 32;
static constexpr int ND      = 10;
static constexpr int DC_DIM  = 16;
static constexpr int NDD     = ND * DC_DIM;          // 160
static constexpr int MTOT    = BATCH * PRIM * PRIM;  // 36864
static constexpr int PPB     = PRIM * PRIM;          // 576

// =====================================================================
// conv1 + BN + ReLU  -> bf16 NHWC activation  h[b][y][x][c]
// Patch staged through LDS with gfx1250 async-to-LDS loads when available.
// =====================================================================
__global__ void __launch_bounds__(256)
conv1_bn_relu(const float* __restrict__ x, const float* __restrict__ w,
              const float* __restrict__ bias, const float* __restrict__ g,
              const float* __restrict__ beta, bf16* __restrict__ h) {
  __shared__ float patch[256];     // 243 used (3*9*9 input taps)
  const int pos = blockIdx.x;      // (b*56+oy)*56+ox
  const int ox = pos % S1;
  const int t1 = pos / S1;
  const int oy = t1 % S1;
  const int b  = t1 / S1;
  const int tid = threadIdx.x;
  if (tid < 243) {
    int ci = tid / 81, rem = tid % 81, ky = rem / 9, kx = rem % 9;
    const float* src = &x[((b * IN_C + ci) * IMG + oy + ky) * IMG + ox + kx];
#if HAS_ASYNC_LDS
    __builtin_amdgcn_global_load_async_to_lds_b32(
        (gint*)src, (lint*)&patch[tid], 0, 0);
#else
    patch[tid] = *src;
#endif
  }
  wait_async_zero();
  __syncthreads();
  const int co = tid;
  const float* wr = w + co * 243;
  float acc = bias[co];
  #pragma unroll 9
  for (int t = 0; t < 243; ++t) acc = fmaf(patch[t], wr[t], acc);
  const float bnscale = 0.99999500003749978f;   // rsqrt(1 + 1e-5)
  acc = acc * g[co] * bnscale + beta[co];
  acc = fmaxf(acc, 0.f);
  h[(size_t)pos * C1 + co] = (bf16)acc;
}

// =====================================================================
// pconv weights -> per-lane B-fragment order, bf16:
//   Wf[tap][kci(8)][ntile(16)][lane(32)][j(16)]
//   lane: n = ntile*16 + (lane&15); k = (j>>3)*16 + (lane>>4)*8 + (j&7); ic = kci*32 + k
// =====================================================================
__global__ void transform_wp(const float* __restrict__ w, bf16* __restrict__ wf) {
  const int idx = blockIdx.x * 256 + threadIdx.x;   // < 81*8*16*32*16 = 81<<16
  const int j   = idx & 15;
  const int l   = (idx >> 4) & 31;
  const int nt  = (idx >> 9) & 15;
  const int kci = (idx >> 13) & 7;
  const int t   = idx >> 16;
  const int ic  = (kci << 5) + ((j >> 3) << 4) + ((l >> 4) << 3) + (j & 7);
  const int oc  = (nt << 4) + (l & 15);
  wf[idx] = (bf16)w[((size_t)oc * 256 + ic) * 81 + t];
}

// =====================================================================
// W_caps[r][n][c][d] f32 -> B-fragment order bf16:
//   Wcf[r][nt(10)][lane(32)][j(16)]
//   N = nt*16 + (lane&15) -> n = nt, d = lane&15 ; K = c = (j>>3)*16 + (lane>>4)*8 + (j&7)
// =====================================================================
__global__ void transform_wc(const float* __restrict__ w, bf16* __restrict__ wcf) {
  const size_t idx = (size_t)blockIdx.x * 256 + threadIdx.x; // < 4608*10*32*16
  const int j = (int)(idx & 15);
  const int l = (int)((idx >> 4) & 31);
  const size_t q = idx >> 9;
  const int nt = (int)(q % ND);
  const int r  = (int)(q / ND);
  const int c  = ((j >> 3) << 4) + ((l >> 4) << 3) + (j & 7);
  const int d  = l & 15;
  wcf[idx] = (bf16)w[(((size_t)r * ND + nt) * PC_DIM + c) * DC_DIM + d];
}

// =====================================================================
// pconv as implicit GEMM, LDS-free, barrier-free.
// M=36864 (b,oy,ox)  N=256 (oc)  K=81*256.
// One wave computes a 16(M) x 64(N) tile: 4 f32 accumulators, each lane
// owns one M row (A frags = native contiguous bf16 NHWC loads), B frags
// are contiguous 32B per lane from the pre-packed Wf.
// =====================================================================
__global__ void __launch_bounds__(256)
pconv_wmma(const bf16* __restrict__ h, const bf16* __restrict__ Wf,
           const float* __restrict__ bias, float* __restrict__ p) {
  const int tid  = threadIdx.x;
  const int lane = tid & 31;
  const int wgid = blockIdx.x * 8 + (tid >> 5);  // 0..9215
  const int mtile = wgid >> 2;                   // 0..2303 (16 rows each)
  const int nt64  = wgid & 3;                    // 0..3 (64 cols each)

  // per-lane A row (fixed for the whole kernel)
  const int mrow = mtile * 16 + (lane & 15);
  const int bA   = mrow / PPB;
  const int remA = mrow % PPB;
  const int oyA  = remA / PRIM, oxA = remA % PRIM;
  const bf16* hrow = h + ((size_t)((bA * S1 + oyA * 2) * S1 + oxA * 2)) * C1
                       + ((lane >> 4) << 3);     // + kb (0 or 8)

  v8f acc[4];
  #pragma unroll
  for (int i = 0; i < 4; ++i)
    #pragma unroll
    for (int j = 0; j < 8; ++j) acc[i][j] = 0.f;

  for (int tap = 0; tap < 81; ++tap) {
    const int ky = tap / 9, kx = tap % 9;
    const bf16* htap = hrow + (ky * S1 + kx) * C1;
    if (tap + 1 < 81) {   // warm next tap's activation line
      const int ky2 = (tap + 1) / 9, kx2 = (tap + 1) % 9;
      __builtin_prefetch(hrow + (ky2 * S1 + kx2) * C1, 0, 1);
    }
    const bf16* wtap = Wf + (((size_t)tap * 8 * 16 + nt64 * 4) * 32 + lane) * 16;
    for (int kci = 0; kci < 8; ++kci) {
      // A fragment: two contiguous 16B loads per lane
      const v8bf lo = *(const v8bf*)(htap + kci * 32);
      const v8bf hi = *(const v8bf*)(htap + kci * 32 + 16);
      v16bf a;
      #pragma unroll
      for (int j = 0; j < 8; ++j) { a[j] = lo[j]; a[8 + j] = hi[j]; }
      const bf16* wk = wtap + (size_t)kci * 16 * 32 * 16;  // advance kci slot
      #pragma unroll
      for (int ns = 0; ns < 4; ++ns) {
        // B fragment: one contiguous 32B load per lane (pre-packed)
        const v16bf bm = *(const v16bf*)(wk + (size_t)ns * 32 * 16);
        acc[ns] = __builtin_amdgcn_wmma_f32_16x16x32_bf16(
            false, a, false, bm, (short)0, acc[ns], false, false);
      }
    }
  }

  // store C tile as p[b][oc][oy][ox] f32 (+ bias), NCHW flat.
  // Per lane the 8 accumulator rows are consecutive rr inside one 16-aligned
  // block (576 % 16 == 0), so use two 16B stores per ns-tile.
  const int mg0 = mtile * 16 + ((lane >> 4) << 3);
  const int bb  = mg0 / PPB, rr0 = mg0 % PPB;
  #pragma unroll
  for (int ns = 0; ns < 4; ++ns) {
    const int oc = nt64 * 64 + ns * 16 + (lane & 15);
    const float bv = bias[oc];
    float* dst = &p[((size_t)(bb * C2 + oc)) * PPB + rr0];
    float4 lo4, hi4;
    lo4.x = acc[ns][0] + bv; lo4.y = acc[ns][1] + bv;
    lo4.z = acc[ns][2] + bv; lo4.w = acc[ns][3] + bv;
    hi4.x = acc[ns][4] + bv; hi4.y = acc[ns][5] + bv;
    hi4.z = acc[ns][6] + bv; hi4.w = acc[ns][7] + bv;
    *(float4*)(dst)     = lo4;
    *(float4*)(dst + 4) = hi4;
  }
}

// =====================================================================
// squash on raw flat view, emit u directly in A-fragment order:
//   Uf[r][mt(4)][lane(32)][j(16)]  with M = batch row
// =====================================================================
__global__ void squash_u(const float* __restrict__ p, bf16* __restrict__ u) {
  const int idx = blockIdx.x * 256 + threadIdx.x;  // b*ROUTES + r
  const int b = idx / ROUTES, r = idx % ROUTES;
  const float* src = p + (size_t)b * (ROUTES * PC_DIM) + r * PC_DIM;
  float v[PC_DIM]; float sn = 0.f;
  #pragma unroll
  for (int c = 0; c < PC_DIM; ++c) { v[c] = src[c]; sn += v[c] * v[c]; }
  const float scale = (sn / (1.f + sn)) * rsqrtf(sn + 1e-8f);
  bf16 sv[PC_DIM];
  #pragma unroll
  for (int c = 0; c < PC_DIM; ++c) sv[c] = (bf16)(v[c] * scale);

  const size_t base = ((size_t)r * 4 + (b >> 4)) * 32;   // lane-group base
  v8bf q0, q1, q2, q3;
  #pragma unroll
  for (int j = 0; j < 8; ++j) {
    q0[j] = sv[j];        // lane b&15,    slots 0..7  : K 0..7
    q1[j] = sv[16 + j];   // lane b&15,    slots 8..15 : K 16..23
    q2[j] = sv[8 + j];    // lane 16+b&15, slots 0..7  : K 8..15
    q3[j] = sv[24 + j];   // lane 16+b&15, slots 8..15 : K 24..31
  }
  bf16* d0 = u + (base + (b & 15)) * 16;
  bf16* d1 = u + (base + 16 + (b & 15)) * 16;
  *(v8bf*)(d0)     = q0;
  *(v8bf*)(d0 + 8) = q1;
  *(v8bf*)(d1)     = q2;
  *(v8bf*)(d1 + 8) = q3;
}

// =====================================================================
// u_hat: per route r, [64x32]x[32x160] -> u_hat[r][64][160] bf16
// LDS-free: A and B fragments are single contiguous 32B loads per lane.
// 5 independent WMMAs per wave (unrolled) for latency overlap.
// =====================================================================
__global__ void __launch_bounds__(256)
uhat_wmma(const bf16* __restrict__ Uf, const bf16* __restrict__ Wcf,
          bf16* __restrict__ uhat) {
  const int r = blockIdx.x;
  const int tid = threadIdx.x;
  const int lane = tid & 31, wave = tid >> 5;
  #pragma unroll
  for (int i = 0; i < 5; ++i) {               // 40 tiles / 8 waves
    const int t = wave + i * 8;
    const int mt = t & 3, nt = t >> 2;
    const v16bf a  = *(const v16bf*)(Uf  + (((size_t)r * 4 + mt) * 32 + lane) * 16);
    const v16bf bm = *(const v16bf*)(Wcf + (((size_t)r * ND + nt) * 32 + lane) * 16);
    v8f acc;
    #pragma unroll
    for (int j = 0; j < 8; ++j) acc[j] = 0.f;
    acc = __builtin_amdgcn_wmma_f32_16x16x32_bf16(
        false, a, false, bm, (short)0, acc, false, false);
    const int n = nt * 16 + (lane & 15);
    #pragma unroll
    for (int k = 0; k < 8; ++k) {
      const int mg = mt * 16 + ((lane >> 4) << 3) + k;
      uhat[((size_t)r * BATCH + mg) * NDD + n] = (bf16)acc[k];
    }
  }
}

// =====================================================================
// dynamic routing pieces
// =====================================================================
__global__ void zero_f32(float* __restrict__ p, int n) {
  int i = blockIdx.x * 256 + threadIdx.x;
  if (i < n) p[i] = 0.f;
}

__global__ void softmax_c(const float* __restrict__ blog, float* __restrict__ c) {
  const size_t idx = (size_t)blockIdx.x * 256 + threadIdx.x;  // b*ROUTES + r
  const float* src = blog + idx * ND;
  float* dst = c + idx * ND;
  float mx = src[0];
  #pragma unroll
  for (int n = 1; n < ND; ++n) mx = fmaxf(mx, src[n]);
  float e[ND]; float s = 0.f;
  #pragma unroll
  for (int n = 0; n < ND; ++n) { e[n] = __expf(src[n] - mx); s += e[n]; }
  const float inv = 1.f / s;
  #pragma unroll
  for (int n = 0; n < ND; ++n) dst[n] = e[n] * inv;
}

// s[b,n,:] = sum_r c[b,r,n]*u_hat[r,b,n,:]; then squash -> v[b,n,:]
__global__ void __launch_bounds__(256)
routing_reduce(const float* __restrict__ c, const bf16* __restrict__ uhat,
               float* __restrict__ v) {
  const int b = blockIdx.x, n = blockIdx.y;
  const int tid = threadIdx.x;
  float part[DC_DIM];
  #pragma unroll
  for (int d = 0; d < DC_DIM; ++d) part[d] = 0.f;
  for (int r = tid; r < ROUTES; r += 256) {
    const float cw = c[((size_t)b * ROUTES + r) * ND + n];
    const bf16* uh = uhat + ((size_t)r * BATCH + b) * NDD + n * DC_DIM;
    #pragma unroll
    for (int d = 0; d < DC_DIM; ++d) part[d] += cw * (float)uh[d];
  }
  __shared__ float red[256 * DC_DIM];
  #pragma unroll
  for (int d = 0; d < DC_DIM; ++d) red[tid * DC_DIM + d] = part[d];
  __syncthreads();
  for (int s2 = 128; s2 > 0; s2 >>= 1) {
    if (tid < s2) {
      #pragma unroll
      for (int d = 0; d < DC_DIM; ++d)
        red[tid * DC_DIM + d] += red[(tid + s2) * DC_DIM + d];
    }
    __syncthreads();
  }
  if (tid == 0) {
    float sn = 0.f;
    #pragma unroll
    for (int d = 0; d < DC_DIM; ++d) sn += red[d] * red[d];
    const float scale = (sn / (1.f + sn)) * rsqrtf(sn + 1e-8f);
    #pragma unroll
    for (int d = 0; d < DC_DIM; ++d)
      v[((size_t)b * ND + n) * DC_DIM + d] = scale * red[d];
  }
}

// b_logits[b,r,n] += sum_d u_hat[r,b,n,d]*v[b,n,d]
__global__ void update_b(const bf16* __restrict__ uhat, const float* __restrict__ v,
                         float* __restrict__ blog) {
  const size_t idx = (size_t)blockIdx.x * 256 + threadIdx.x;  // b*ROUTES + r
  const int b = (int)(idx / ROUTES), r = (int)(idx % ROUTES);
  const bf16* uh = uhat + ((size_t)r * BATCH + b) * NDD;
  const float* vb = v + (size_t)b * NDD;
  float* dst = blog + idx * ND;
  #pragma unroll
  for (int n = 0; n < ND; ++n) {
    float dot = 0.f;
    #pragma unroll
    for (int d = 0; d < DC_DIM; ++d)
      dot += (float)uh[n * DC_DIM + d] * vb[n * DC_DIM + d];
    dst[n] += dot;
  }
}

// =====================================================================
// tiny FC head (f32): out[m,n] = relu?(in[m,:]·w[n,:] + bias[n])
// =====================================================================
__global__ void fc_kernel(const float* __restrict__ in, const float* __restrict__ w,
                          const float* __restrict__ bias, float* __restrict__ out,
                          int M, int N, int K, int relu) {
  const int idx = blockIdx.x * 256 + threadIdx.x;
  if (idx >= M * N) return;
  const int m = idx / N, n = idx % N;
  const float* a  = in + (size_t)m * K;
  const float* ww = w + (size_t)n * K;
  float acc = bias[n];
  for (int k = 0; k < K; ++k) acc = fmaf(a[k], ww[k], acc);
  if (relu) acc = fmaxf(acc, 0.f);
  out[idx] = acc;
}

// =====================================================================
// host side
// =====================================================================
extern "C" void kernel_launch(void* const* d_in, const int* in_sizes, int n_in,
                              void* d_out, int out_size, void* d_ws, size_t ws_size,
                              hipStream_t stream) {
  (void)in_sizes; (void)n_in; (void)out_size; (void)ws_size;
  const float* x       = (const float*)d_in[0];
  const float* conv1_w = (const float*)d_in[1];
  const float* conv1_b = (const float*)d_in[2];
  const float* bn_g    = (const float*)d_in[3];
  const float* bn_b    = (const float*)d_in[4];
  const float* pconv_w = (const float*)d_in[5];
  const float* pconv_b = (const float*)d_in[6];
  const float* W_caps  = (const float*)d_in[7];
  const float* fc1_w   = (const float*)d_in[8];
  const float* fc1_b   = (const float*)d_in[9];
  const float* fc2_w   = (const float*)d_in[10];
  const float* fc2_b   = (const float*)d_in[11];
  const float* fc3_w   = (const float*)d_in[12];
  const float* fc3_b   = (const float*)d_in[13];

  char* ws = (char*)d_ws;
  size_t off = 0;
  auto carve = [&](size_t bytes) -> char* {
    char* pp = ws + off;
    off = (off + bytes + 255) & ~(size_t)255;
    return pp;
  };
  bf16*  h    = (bf16*) carve((size_t)BATCH * S1 * S1 * C1 * sizeof(bf16));
  bf16*  Wf   = (bf16*) carve((size_t)81 * C1 * C2 * sizeof(bf16));
  float* p    = (float*)carve((size_t)BATCH * ROUTES * PC_DIM * sizeof(float));
  bf16*  Uf   = (bf16*) carve((size_t)ROUTES * BATCH * PC_DIM * sizeof(bf16));
  bf16*  Wcf  = (bf16*) carve((size_t)ROUTES * PC_DIM * NDD * sizeof(bf16));
  bf16*  uhat = (bf16*) carve((size_t)ROUTES * BATCH * NDD * sizeof(bf16));
  float* blog = (float*)carve((size_t)BATCH * ROUTES * ND * sizeof(float));
  float* cbuf = (float*)carve((size_t)BATCH * ROUTES * ND * sizeof(float));
  float* v    = (float*)carve((size_t)BATCH * NDD * sizeof(float));
  float* f1   = (float*)carve((size_t)BATCH * 512 * sizeof(float));
  float* f2   = (float*)carve((size_t)BATCH * 256 * sizeof(float));

  // 1) conv1 + BN + relu -> bf16 NHWC (async-to-LDS staging when available)
  conv1_bn_relu<<<BATCH * S1 * S1, 256, 0, stream>>>(x, conv1_w, conv1_b, bn_g, bn_b, h);
  // 2) weight repacks into per-lane WMMA fragment order
  transform_wp<<<(81 * C1 * C2) / 256, 256, 0, stream>>>(pconv_w, Wf);
  transform_wc<<<(ROUTES * PC_DIM * NDD) / 256, 256, 0, stream>>>(W_caps, Wcf);
  // 3) pconv implicit GEMM (bf16 WMMA, f32 accum, LDS-free)
  pconv_wmma<<<(MTOT / 16) * (C2 / 64) / 8, 256, 0, stream>>>(h, Wf, pconv_b, p);
  // 4) squash -> Uf (bf16, A-fragment order)
  squash_u<<<(BATCH * ROUTES) / 256, 256, 0, stream>>>(p, Uf);
  // 5) u_hat per-route GEMMs (bf16 WMMA, LDS-free)
  uhat_wmma<<<ROUTES, 256, 0, stream>>>(Uf, Wcf, uhat);
  // 6) 3-iteration dynamic routing
  const int nblog = BATCH * ROUTES * ND;
  zero_f32<<<(nblog + 255) / 256, 256, 0, stream>>>(blog, nblog);
  for (int it = 0; it < 3; ++it) {
    softmax_c<<<(BATCH * ROUTES) / 256, 256, 0, stream>>>(blog, cbuf);
    routing_reduce<<<dim3(BATCH, ND), 256, 0, stream>>>(cbuf, uhat, v);
    if (it < 2)
      update_b<<<(BATCH * ROUTES) / 256, 256, 0, stream>>>(uhat, v, blog);
  }
  // 7) FC head
  fc_kernel<<<(BATCH * 512) / 256, 256, 0, stream>>>(v,  fc1_w, fc1_b, f1, BATCH, 512, NDD, 1);
  fc_kernel<<<(BATCH * 256) / 256, 256, 0, stream>>>(f1, fc2_w, fc2_b, f2, BATCH, 256, 512, 1);
  fc_kernel<<<(BATCH * 128) / 256, 256, 0, stream>>>(f2, fc3_w, fc3_b, (float*)d_out, BATCH, 128, 256, 0);
}
